// SelfAttention_67525475827753
// MI455X (gfx1250) — matrix operand order repaired and checked by reading, not compiled, not run on hardware
//
#include <hip/hip_runtime.h>
#include <hip/hip_bf16.h>

typedef __bf16 v16bf __attribute__((ext_vector_type(16)));
typedef __bf16 v8bf  __attribute__((ext_vector_type(8)));
typedef float  v8f   __attribute__((ext_vector_type(8)));

#define CC 256
#define NN 4096
#define BB 4
#define HEADS 4
#define HD 64

__device__ __forceinline__ v8f zero8() {
  v8f z;
#pragma unroll
  for (int i = 0; i < 8; ++i) z[i] = 0.f;
  return z;
}

__device__ __forceinline__ v8f wmma_bf16(v16bf a, v16bf b, v8f c) {
  // D = A(16x32 bf16) * B(32x16 bf16) + C(16x16 f32)
  return __builtin_amdgcn_wmma_f32_16x16x32_bf16(false, a, false, b, (short)0, c,
                                                 false, false);
}

// A-operand (16x32, 16-bit): lane holds row M = lane%16.
// element e <-> K = (e&7) + 16*(e>>3) + 8*(lane>=16).
// Caller passes p = row_base + kchunk + hi*8; elements 0..7 = p[0..7],
// elements 8..15 = p[16..23]  (two contiguous 16B loads).
__device__ __forceinline__ v16bf load_a16(const __bf16* p) {
  v8bf r0 = *(const v8bf*)(p);
  v8bf r1 = *(const v8bf*)(p + 16);
  v16bf a;
#pragma unroll
  for (int i = 0; i < 8; ++i) { a[i] = r0[i]; a[i + 8] = r1[i]; }
  return a;
}

__device__ __forceinline__ float redmax16(float v) {
#pragma unroll
  for (int m = 1; m <= 8; m <<= 1) v = fmaxf(v, __shfl_xor(v, m, 32));
  return v;
}
__device__ __forceinline__ float redsum16(float v) {
#pragma unroll
  for (int m = 1; m <= 8; m <<= 1) v += __shfl_xor(v, m, 32);
  return v;
}

// CDNA5 async global->LDS copy (bypasses VGPRs; tracked by ASYNCcnt).
// Inline asm per bridge doc: builtin arity differs across toolchains.
__device__ __forceinline__ void async_copy_b128(unsigned lds_byte_addr,
                                                unsigned long long gaddr) {
  asm volatile("global_load_async_to_lds_b128 %0, %1, off"
               :: "v"(lds_byte_addr), "v"(gaddr)
               : "memory");
}
__device__ __forceinline__ void wait_async0() {
  asm volatile("s_wait_asynccnt 0" ::: "memory");
}

// ---------------- Kernel 1: LayerNorm -> bf16 tokens [B*N, C] ----------------
__global__ __launch_bounds__(256) void ln_kernel(const float* __restrict__ x,
                                                 const float* __restrict__ gamma,
                                                 const float* __restrict__ beta,
                                                 __bf16* __restrict__ xn) {
  const int t = blockIdx.x * 256 + threadIdx.x;   // token id 0..16383
  const int b = t >> 12, pix = t & 4095;
  const float* xb = x + (size_t)b * CC * NN + pix;
  float s = 0.f, ss = 0.f;
  for (int c = 0; c < CC; ++c) {
    float v = xb[(size_t)c * NN];
    s += v; ss += v * v;
  }
  const float mu = s * (1.f / CC);
  const float var = ss * (1.f / CC) - mu * mu;
  const float rs = rsqrtf(var + 1e-5f);
  __bf16* o = xn + (size_t)t * CC;
  for (int c = 0; c < CC; ++c) {
    float v = (xb[(size_t)c * NN] - mu) * rs * gamma[c] + beta[c];
    o[c] = (__bf16)v;
  }
}

// ------------- Kernel 2: transpose+convert weights to bf16 W^T ---------------
__global__ __launch_bounds__(256) void prep_w_kernel(const float* __restrict__ wqkv,
                                                     const float* __restrict__ wproj,
                                                     __bf16* __restrict__ wqkvT,
                                                     __bf16* __restrict__ wprojT) {
  const int i = blockIdx.x * 256 + threadIdx.x;   // 0..262143
  if (i < 768 * 256) {
    const int j = i / 256, c = i % 256;
    wqkvT[i] = (__bf16)wqkv[c * 768 + j];
  } else {
    const int i2 = i - 768 * 256;                 // 0..65535
    const int j = i2 / 256, c = i2 % 256;
    wprojT[i2] = (__bf16)wproj[c * 256 + j];
  }
}

// -------- Kernel 3: QKV GEMM (WMMA), scatter to Q/K [b,h,tok,hd], V^T --------
__global__ __launch_bounds__(256) void qkv_kernel(const __bf16* __restrict__ xn,
                                                  const __bf16* __restrict__ wT,
                                                  const float* __restrict__ bias,
                                                  __bf16* __restrict__ qbuf,
                                                  __bf16* __restrict__ kbuf,
                                                  __bf16* __restrict__ vT) {
  const int tid = threadIdx.x;
  const int wave = tid >> 5, lane = tid & 31, lo = lane & 15, hi = lane >> 4;
  const int mtile = blockIdx.x;                 // 0..1023 (rows of 16)
  const int ntile = blockIdx.y * 8 + wave;      // 0..47
  const int j0 = ntile * 16;
  v8f acc = zero8();
  const __bf16* arow = xn + (size_t)(mtile * 16 + lo) * CC;
  const __bf16* brow = wT + (size_t)(j0 + lo) * CC;
#pragma unroll
  for (int kc = 0; kc < CC; kc += 32) {
    v16bf a = load_a16(arow + kc + hi * 8);
    v16bf bm = load_a16(brow + kc + hi * 8);
    acc = wmma_bf16(a, bm, acc);
  }
  const int j = j0 + lo;
  const float bj = bias[j];
#pragma unroll
  for (int r = 0; r < 8; ++r) {
    const int mrow = mtile * 16 + r + 8 * hi;
    const int b = mrow >> 12, tok = mrow & 4095;
    const float val = acc[r] + bj;
    if (j0 < 256) {
      const int h = j >> 6, hd = j & 63;
      qbuf[((size_t)((b * HEADS + h) * NN) + tok) * HD + hd] = (__bf16)val;
    } else if (j0 < 512) {
      const int jj = j - 256, h = jj >> 6, hd = jj & 63;
      kbuf[((size_t)((b * HEADS + h) * NN) + tok) * HD + hd] = (__bf16)val;
    } else {
      const int jj = j - 512, h = jj >> 6, hd = jj & 63;
      vT[((size_t)((b * HEADS + h) * HD) + hd) * NN + tok] = (__bf16)val;
    }
  }
}

// ------- Kernel 4: flash attention, WMMA + double-buffered async LDS ---------
__global__ __launch_bounds__(256) void attn_kernel(const __bf16* __restrict__ q,
                                                   const __bf16* __restrict__ k,
                                                   const __bf16* __restrict__ vT,
                                                   __bf16* __restrict__ attnout) {
  __shared__ __bf16 ks[2][64 * 64];     // [key][hd]      2 x 8 KB
  __shared__ __bf16 vs[2][64 * 64];     // [hd][key]      2 x 8 KB (V^T tile)
  __shared__ __bf16 ps[8][16 * 64];     // per-wave P     16 KB
  const int tid = threadIdx.x;
  const int wave = tid >> 5, lane = tid & 31, lo = lane & 15, hi = lane >> 4;
  const int b = blockIdx.z, h = blockIdx.y;
  const int q0 = blockIdx.x * 128 + wave * 16;
  const __bf16* qb = q + (size_t)((b * HEADS + h) * NN) * HD;
  const __bf16* kb = k + (size_t)((b * HEADS + h) * NN) * HD;
  const __bf16* vb = vT + (size_t)((b * HEADS + h) * HD) * NN;

  // Q tile 16x64 held as two A operands (K = 0..31, 32..63)
  const v16bf aq0 = load_a16(qb + (size_t)(q0 + lo) * HD + hi * 8);
  const v16bf aq1 = load_a16(qb + (size_t)(q0 + lo) * HD + 32 + hi * 8);

  // async stage of one 64-key chunk into buffer bufi
  auto stage = [&](int kc, int bufi) {
    // K tile: 64 keys x 64 hd, fully contiguous 8 KB -> 512 b128, 2 per thread
    const unsigned ldsk = (unsigned)(size_t)(&ks[bufi][0]);
    const unsigned long long gk = (unsigned long long)(size_t)(kb + (size_t)kc * HD);
    async_copy_b128(ldsk + (unsigned)tid * 16, gk + (unsigned long long)tid * 16);
    async_copy_b128(ldsk + 4096u + (unsigned)tid * 16,
                    gk + 4096ull + (unsigned long long)tid * 16);
    // V^T tile: 64 hd-rows x 64 keys, row stride NN -> 2 b128 per thread
    const int row = tid >> 2, c0 = (tid & 3) * 16;
    const unsigned ldsv =
        (unsigned)(size_t)(&vs[bufi][0]) + (unsigned)(row * 64 + c0) * 2u;
    const unsigned long long gv =
        (unsigned long long)(size_t)(vb + (size_t)row * NN + kc + c0);
    async_copy_b128(ldsv, gv);
    async_copy_b128(ldsv + 16u, gv + 16ull);
  };

  v8f acc[4];
#pragma unroll
  for (int t = 0; t < 4; ++t) acc[t] = zero8();
  float m[8], l[8];
#pragma unroll
  for (int r = 0; r < 8; ++r) { m[r] = -1e30f; l[r] = 0.f; }
  const float scale = 0.0625f;  // 1/sqrt(256)

  stage(0, 0);
  int buf = 0;
  for (int kc = 0; kc < NN; kc += 64) {
    wait_async0();        // own wave's async copies into `buf` complete
    __syncthreads();      // everyone's copies done; prev reads of buf^1 done
    if (kc + 64 < NN) stage(kc + 64, buf ^ 1);   // prefetch next chunk
    const __bf16* kst = &ks[buf][0];
    const __bf16* vst = &vs[buf][0];

    // ---- scores: 4 tiles of 16 keys (D layout: col=key, row=r+8*hi) ----
    float p[4][8];
    float rowm[8];
#pragma unroll
    for (int r = 0; r < 8; ++r) rowm[r] = -1e30f;
#pragma unroll
    for (int t = 0; t < 4; ++t) {
      v8f s = zero8();
      v16bf b0 = load_a16(kst + (t * 16 + lo) * 64 + hi * 8);
      s = wmma_bf16(aq0, b0, s);
      v16bf b1 = load_a16(kst + (t * 16 + lo) * 64 + 32 + hi * 8);
      s = wmma_bf16(aq1, b1, s);
#pragma unroll
      for (int r = 0; r < 8; ++r) {
        p[t][r] = s[r] * scale;
        rowm[r] = fmaxf(rowm[r], p[t][r]);
      }
    }
    // ---- online softmax: row stats across the 16 key-lanes of each half ----
    float fs[8], lsum[8];
#pragma unroll
    for (int r = 0; r < 8; ++r) {
      rowm[r] = redmax16(rowm[r]);
      const float mn = fmaxf(m[r], rowm[r]);
      fs[r] = __expf(m[r] - mn);
      m[r] = mn;
      lsum[r] = 0.f;
    }
#pragma unroll
    for (int t = 0; t < 4; ++t)
#pragma unroll
      for (int r = 0; r < 8; ++r) {
        p[t][r] = __expf(p[t][r] - m[r]);
        lsum[r] += p[t][r];
      }
#pragma unroll
    for (int r = 0; r < 8; ++r) {
      lsum[r] = redsum16(lsum[r]);
      l[r] = l[r] * fs[r] + lsum[r];
    }
#pragma unroll
    for (int t = 0; t < 4; ++t)
#pragma unroll
      for (int r = 0; r < 8; ++r) acc[t][r] *= fs[r];

    // ---- transpose P (D layout -> A layout) through per-wave LDS ----
    __bf16* pw = ps[wave];
#pragma unroll
    for (int t = 0; t < 4; ++t)
#pragma unroll
      for (int r = 0; r < 8; ++r)
        pw[(r + 8 * hi) * 64 + t * 16 + lo] = (__bf16)p[t][r];
    asm volatile("" ::: "memory");   // keep DS store->load ordered (in-wave, in-order LDS)
    const v16bf ap0 = load_a16(pw + lo * 64 + hi * 8);
    const v16bf ap1 = load_a16(pw + lo * 64 + 32 + hi * 8);

    // ---- out += P(16x64) * V(64x16) per 16-wide hd tile ----
#pragma unroll
    for (int ht = 0; ht < 4; ++ht) {
      v16bf bv0 = load_a16(vst + (ht * 16 + lo) * 64 + hi * 8);
      acc[ht] = wmma_bf16(ap0, bv0, acc[ht]);
      v16bf bv1 = load_a16(vst + (ht * 16 + lo) * 64 + 32 + hi * 8);
      acc[ht] = wmma_bf16(ap1, bv1, acc[ht]);
    }
    buf ^= 1;
  }

  float inv[8];
#pragma unroll
  for (int r = 0; r < 8; ++r) inv[r] = 1.f / l[r];
#pragma unroll
  for (int ht = 0; ht < 4; ++ht)
#pragma unroll
    for (int r = 0; r < 8; ++r) {
      const size_t row = (size_t)b * NN + q0 + r + 8 * hi;
      attnout[row * CC + h * HD + ht * 16 + lo] = (__bf16)(acc[ht][r] * inv[r]);
    }
}

// --------- Kernel 5: proj GEMM (WMMA) + bias + residual -> [B,C,H,W] ---------
__global__ __launch_bounds__(256) void proj_kernel(const __bf16* __restrict__ attn,
                                                   const __bf16* __restrict__ wT,
                                                   const float* __restrict__ bias,
                                                   const float* __restrict__ x,
                                                   float* __restrict__ out) {
  const int tid = threadIdx.x;
  const int wave = tid >> 5, lane = tid & 31, lo = lane & 15, hi = lane >> 4;
  const int mtile = blockIdx.x;              // 0..1023
  const int ntile = blockIdx.y * 8 + wave;   // 0..15
  const int c0 = ntile * 16;
  v8f acc = zero8();
  const __bf16* arow = attn + (size_t)(mtile * 16 + lo) * CC;
  const __bf16* brow = wT + (size_t)(c0 + lo) * CC;
#pragma unroll
  for (int kc = 0; kc < CC; kc += 32) {
    v16bf a = load_a16(arow + kc + hi * 8);
    v16bf bm = load_a16(brow + kc + hi * 8);
    acc = wmma_bf16(a, bm, acc);
  }
  const int cout = c0 + lo;
  const float bc = bias[cout];
#pragma unroll
  for (int r = 0; r < 8; ++r) {
    const int mrow = mtile * 16 + r + 8 * hi;
    const int b = mrow >> 12, pix = mrow & 4095;
    const size_t idx = ((size_t)(b * CC + cout)) * NN + pix;
    out[idx] = acc[r] + bc + x[idx];
  }
}

extern "C" void kernel_launch(void* const* d_in, const int* in_sizes, int n_in,
                              void* d_out, int out_size, void* d_ws, size_t ws_size,
                              hipStream_t stream) {
  (void)in_sizes; (void)n_in; (void)out_size; (void)ws_size;
  const float* x     = (const float*)d_in[0];
  const float* gam   = (const float*)d_in[1];
  const float* bet   = (const float*)d_in[2];
  const float* wqkv  = (const float*)d_in[3];
  const float* bqkv  = (const float*)d_in[4];
  const float* wproj = (const float*)d_in[5];
  const float* bproj = (const float*)d_in[6];
  float* out = (float*)d_out;

  char* ws = (char*)d_ws;
  __bf16* xn     = (__bf16*)(ws);                       //  8 MB  [16384,256]
  __bf16* qbuf   = (__bf16*)(ws + (8u << 20));          //  8 MB  [b,h,tok,hd]
  __bf16* kbuf   = (__bf16*)(ws + (16u << 20));         //  8 MB  [b,h,tok,hd]
  __bf16* vTbuf  = (__bf16*)(ws + (24u << 20));         //  8 MB  [b,h,hd,tok]
  __bf16* attn   = (__bf16*)(ws + (32u << 20));         //  8 MB  [16384,256]
  __bf16* wqkvT  = (__bf16*)(ws + (40u << 20));         //  384 KB [768,256]
  __bf16* wprojT = (__bf16*)(ws + (40u << 20) + 393216);//  128 KB [256,256]

  ln_kernel<<<64, 256, 0, stream>>>(x, gam, bet, xn);
  prep_w_kernel<<<1024, 256, 0, stream>>>(wqkv, wproj, wqkvT, wprojT);
  qkv_kernel<<<dim3(1024, 6), 256, 0, stream>>>(xn, wqkvT, bqkv, qbuf, kbuf, vTbuf);
  attn_kernel<<<dim3(NN / 128, HEADS, BB), 256, 0, stream>>>(qbuf, kbuf, vTbuf, attn);
  proj_kernel<<<dim3(1024, 2), 256, 0, stream>>>(attn, wprojT, bproj, x, out);
}
